// SeqPropHead_38465727103466
// MI455X (gfx1250) — compile-verified, matrix operand order checked by reading, not run on hardware
//
#include <hip/hip_runtime.h>
#include <hip/hip_bf16.h>
#include <cstdint>
#include <cmath>

typedef __attribute__((ext_vector_type(16))) __bf16 v16bf;
typedef __attribute__((ext_vector_type(8)))  __bf16 v8bf;
typedef __attribute__((ext_vector_type(2)))  __bf16 v2bf;
typedef __attribute__((ext_vector_type(8)))  float  v8f;
typedef __attribute__((ext_vector_type(4)))  float  v4f;

#define SHFL16(lo, hi) __builtin_shufflevector(lo, hi, 0,1,2,3,4,5,6,7,8,9,10,11,12,13,14,15)

// LDS tile row stride: 40 bf16 = 80 bytes (16B aligned rows, banks spread).
#define LDSK 40

// ---------------------------------------------------------------------------
// Weight pre-conversion: f32 OIHW (Cout, Ktot) -> bf16 (Cout, Kpad), K zero-
// padded to a multiple of 32 so the conv inner loop needs no k-masking and
// the B tile can be staged with raw async byte copies.
// ---------------------------------------------------------------------------
__global__ void wcvt_kernel(const float* __restrict__ w, __bf16* __restrict__ o,
                            int Cout, int Ktot, int Kpad)
{
    const int64_t idx = (int64_t)blockIdx.x * blockDim.x + threadIdx.x;
    const int64_t total = (int64_t)Cout * Kpad;
    if (idx >= total) return;
    const int n = (int)(idx / Kpad);
    const int k = (int)(idx - (int64_t)n * Kpad);
    o[idx] = (k < Ktot) ? (__bf16)w[(int64_t)n * Ktot + k] : (__bf16)0.0f;
}

// ---------------------------------------------------------------------------
// Implicit-GEMM convolution with WMMA (bf16 in, f32 accumulate), LDS-staged,
// double-buffered. GEMM view: M = B*OH*OW, N = Cout, K = Cin*KH*KW -> Kpad.
// Block = 256 threads = 8 waves. Block tile: 256 M x 64 N, K chunk = 32.
// Wave tile: 32 M x 64 N (2 A-frags x 4 B-frags -> 8 WMMA / chunk).
//  - A tile (256x32): coalesced f32 loads, im2col gather, branch-free border
//    masking + fused input-ReLU, bf16-packed LDS stores.
//  - B tile (64x32): raw async global->LDS b64 copies of preconverted,
//    zero-padded bf16 weights (ASYNCcnt; s_wait_asynccnt).
//  - Ping-pong LDS buffers: one barrier + one asynccnt-wait per chunk;
//    next-chunk staging overlaps current-chunk ds_load_b128 + WMMA.
// Epilogue: hoisted run guards + b128 vector loads/stores (8-aligned runs),
// fused bias + optional residual (optionally batch-broadcast).
// ---------------------------------------------------------------------------
template<int KH, int KW>
__global__ __launch_bounds__(256) void conv_wmma_kernel(
    const float* __restrict__ in, const __bf16* __restrict__ wgt,
    const float* __restrict__ bias, const float* res, float* out,
    int B, int Cin, int IH, int IW, int Cout, int OH, int OW,
    int stride, int pad, int relu_in, int res_bcast)
{
    constexpr int KHW = KH * KW;
    __shared__ __bf16 Alds[2][256][LDSK];
    __shared__ __bf16 Blds[2][64][LDSK];

    const int tid  = threadIdx.x;
    const int lane = tid & 31;
    const int wave = tid >> 5;
    const int ohw  = OH * OW;
    const int Mtot = B * ohw;
    const int Ktot = Cin * KHW;
    const int Kpad = (Ktot + 31) & ~31;
    const int m_blk  = blockIdx.x * 256;
    const int n_base = blockIdx.y * 64;

    // ---- A-stage decode: one output position (row) per thread, all 32 k ----
    const int ml = tid;
    int mg = m_blk + ml; if (mg >= Mtot) mg = Mtot - 1;
    const int bL   = mg / ohw;
    const int remL = mg - bL * ohw;
    const int ohL  = remL / OW;
    const int owL  = remL - ohL * OW;
    const int ih0  = ohL * stride - pad;
    const int iw0  = owL * stride - pad;
    const float* inb = in + (int64_t)bL * Cin * IH * IW;

    // ---- B-stage decode: 2 async b64 copies per thread ----
    const int kq = (tid & 7) * 4;               // 0..28
    const int nb = tid >> 3;                    // 0..31

    // ---- fragment addressing ----
    const int rowA0 = wave * 32 + (lane & 15);  // first A subtile row
    const int klo   = (lane >> 4) * 8;          // element offset within chunk

    auto stageB = [&](int kc, int buf) {
#pragma unroll
        for (int i = 0; i < 2; ++i) {
            const int nl = nb + 32 * i;
            int n = n_base + nl; if (n >= Cout) n = Cout - 1;
            const unsigned ldsoff = (unsigned)(uintptr_t)&Blds[buf][nl][kq];
            const unsigned goff   = (unsigned)(((int64_t)n * Kpad + kc + kq) * 2);
            asm volatile("global_load_async_to_lds_b64 %0, %1, %2 offset:0"
                         :: "v"(ldsoff), "v"(goff), "s"(wgt) : "memory");
        }
    };
    auto stageA = [&](int kc, int buf) {
#pragma unroll
        for (int i = 0; i < 16; ++i) {
            float v01[2];
#pragma unroll
            for (int u = 0; u < 2; ++u) {
                const int k = kc + 2 * i + u;
                const int kcl = k < Ktot ? k : Ktot - 1;   // address clamp only
                const int ci = kcl / KHW;
                const int r  = kcl - ci * KHW;
                const int ky = r / KW;
                const int kx = r - ky * KW;
                const int ih = ih0 + ky;
                const int iw = iw0 + kx;
                const int ihc = ih < 0 ? 0 : (ih >= IH ? IH - 1 : ih);
                const int iwc = iw < 0 ? 0 : (iw >= IW ? IW - 1 : iw);
                float t = inb[((int64_t)ci * IH + ihc) * IW + iwc];
                if (relu_in && t < 0.0f) t = 0.0f;
                const bool ok = ((unsigned)ih < (unsigned)IH) &
                                ((unsigned)iw < (unsigned)IW);
                v01[u] = ok ? t : 0.0f;   // k >= Ktot taps hit zero weights
            }
            v2bf pk;
            pk[0] = (__bf16)v01[0];
            pk[1] = (__bf16)v01[1];
            *(v2bf*)&Alds[buf][ml][2 * i] = pk;
        }
    };

    v8f acc[2][4] = {};

    const int nch = Kpad >> 5;
    stageB(0, 0);
    stageA(0, 0);
    for (int it = 0; it < nch; ++it) {
        asm volatile("s_wait_asynccnt 0x0" ::: "memory");
        __syncthreads();
        const int cur = it & 1;
        if (it + 1 < nch) {                      // stage next chunk (overlaps WMMA)
            stageB((it + 1) * 32, cur ^ 1);
            stageA((it + 1) * 32, cur ^ 1);
        }
        // ---------- fragments + WMMA (2 M-subtiles x 4 N-subtiles) ----------
        v16bf af[2];
#pragma unroll
        for (int s = 0; s < 2; ++s) {
            const v8bf a_lo = *(const v8bf*)&Alds[cur][rowA0 + 16 * s][klo];
            const v8bf a_hi = *(const v8bf*)&Alds[cur][rowA0 + 16 * s][klo + 16];
            af[s] = SHFL16(a_lo, a_hi);
        }
#pragma unroll
        for (int j = 0; j < 4; ++j) {
            const int rowB = j * 16 + (lane & 15);
            const v8bf b_lo = *(const v8bf*)&Blds[cur][rowB][klo];
            const v8bf b_hi = *(const v8bf*)&Blds[cur][rowB][klo + 16];
            const v16bf bfrag = SHFL16(b_lo, b_hi);
#pragma unroll
            for (int s = 0; s < 2; ++s) {
                acc[s][j] = __builtin_amdgcn_wmma_f32_16x16x32_bf16(
                    false, af[s], false, bfrag, (short)0, acc[s][j], false, false);
            }
        }
    }

    // ---------- epilogue: 8-run vector stores (ohw & Mtot are multiples of 8,
    // so an aligned 8-run never crosses a batch boundary) ----------
    const int ncol = lane & 15;
#pragma unroll
    for (int s = 0; s < 2; ++s) {
        const int mrow0 = m_blk + wave * 32 + 16 * s + 8 * (lane >> 4);
        if (mrow0 >= Mtot) continue;
        const int b   = mrow0 / ohw;
        const int rem = mrow0 - b * ohw;
#pragma unroll
        for (int j = 0; j < 4; ++j) {
            const int n = n_base + j * 16 + ncol;
            if (n >= Cout) continue;
            const float bv = bias[n];
            const int64_t base = ((int64_t)b * Cout + n) * ohw + rem;
            v4f o0, o1;
#pragma unroll
            for (int r = 0; r < 4; ++r) {
                o0[r] = acc[s][j][r] + bv;
                o1[r] = acc[s][j][r + 4] + bv;
            }
            if (res) {
                const int rb = res_bcast ? 0 : b;
                const int64_t rbase = ((int64_t)rb * Cout + n) * ohw + rem;
                const v4f r0 = *(const v4f*)&res[rbase];
                const v4f r1 = *(const v4f*)&res[rbase + 4];
#pragma unroll
                for (int r = 0; r < 4; ++r) { o0[r] += r0[r]; o1[r] += r1[r]; }
            }
            *(v4f*)&out[base]     = o0;
            *(v4f*)&out[base + 4] = o1;
        }
    }
}

// ---------------------------------------------------------------------------
// Generic strided batched GEMM with WMMA, LDS-staged:
//   C[z][m][n] = scale * sum_k A[z*a_batch + m*a_ms + k*a_ks]
//                             * B[z*b_batch + k*b_ks + n*b_ns]
//   C element: C[z*c_batch + m*N + n]
// Block tile 128 M x 64 N, wave tile 16 M x 64 N.
// ---------------------------------------------------------------------------
__global__ __launch_bounds__(256) void gemm_wmma_kernel(
    const float* __restrict__ A, const float* __restrict__ Bm, float* __restrict__ C,
    int M, int N, int K,
    long long a_ms, long long a_ks, long long b_ks, long long b_ns,
    long long a_batch, long long b_batch, long long c_batch, float scale)
{
    __shared__ __bf16 Alds[128][LDSK];
    __shared__ __bf16 Blds[64][LDSK];

    const int tid  = threadIdx.x;
    const int lane = tid & 31;
    const int wave = tid >> 5;
    const float* Ab = A + (int64_t)blockIdx.z * a_batch;
    const float* Bb = Bm + (int64_t)blockIdx.z * b_batch;
    float*       Cb = C + (int64_t)blockIdx.z * c_batch;
    const int m_blk  = blockIdx.x * 128;
    const int n_base = blockIdx.y * 64;

    const int ml    = tid & 127;
    const int khalf = tid >> 7;
    int mg = m_blk + ml; if (mg >= M) mg = M - 1;
    const float* Arow = Ab + (int64_t)mg * a_ms;

    const int kB  = tid & 31;
    const int nB0 = tid >> 5;

    const int rowA = wave * 16 + (lane & 15);
    const int klo  = (lane >> 4) * 8;

    v8f acc[4] = {};

    for (int kc = 0; kc < K; kc += 32) {
        __syncthreads();
#pragma unroll
        for (int i = 0; i < 8; ++i) {
            const int kbase = kc + khalf * 16 + 2 * i;
            float v01[2];
#pragma unroll
            for (int u = 0; u < 2; ++u) {
                const int k = kbase + u;
                const int kcl = k < K ? k : K - 1;
                float t = Arow[(int64_t)kcl * a_ks];
                v01[u] = (k < K) ? t : 0.0f;
            }
            v2bf pk;
            pk[0] = (__bf16)v01[0];
            pk[1] = (__bf16)v01[1];
            *(v2bf*)&Alds[ml][khalf * 16 + 2 * i] = pk;
        }
#pragma unroll
        for (int i = 0; i < 8; ++i) {
            const int nl = nB0 + 8 * i;
            int n = n_base + nl; if (n >= N) n = N - 1;
            const int k = kc + kB;
            const int kcl = k < K ? k : K - 1;
            float v = Bb[(int64_t)kcl * b_ks + (int64_t)n * b_ns];
            v = (k < K) ? v : 0.0f;
            Blds[nl][kB] = (__bf16)v;
        }
        __syncthreads();
        const v8bf a_lo = *(const v8bf*)&Alds[rowA][klo];
        const v8bf a_hi = *(const v8bf*)&Alds[rowA][klo + 16];
        const v16bf afrag = SHFL16(a_lo, a_hi);
#pragma unroll
        for (int j = 0; j < 4; ++j) {
            const int rowB = j * 16 + (lane & 15);
            const v8bf b_lo = *(const v8bf*)&Blds[rowB][klo];
            const v8bf b_hi = *(const v8bf*)&Blds[rowB][klo + 16];
            const v16bf bfrag = SHFL16(b_lo, b_hi);
            acc[j] = __builtin_amdgcn_wmma_f32_16x16x32_bf16(
                false, afrag, false, bfrag, (short)0, acc[j], false, false);
        }
    }

    const int mrow0 = m_blk + wave * 16 + 8 * (lane >> 4);
    const int ncol  = lane & 15;
    if (mrow0 < M) {                       // M is a multiple of 8: run all-valid
#pragma unroll
        for (int j = 0; j < 4; ++j) {
            const int n = n_base + j * 16 + ncol;
            if (n >= N) continue;
#pragma unroll
            for (int r = 0; r < 8; ++r) {
                Cb[(int64_t)(mrow0 + r) * N + n] = acc[j][r] * scale;
            }
        }
    }
}

// Softmax over the M axis of att[z][m][n] (column softmax). One block per (n,z).
__global__ __launch_bounds__(256) void softmax_col_kernel(float* att, int M, int N)
{
    __shared__ float sred[256];
    const int n = blockIdx.x;
    float* base = att + (int64_t)blockIdx.y * M * N + n;
    const int tid = threadIdx.x;
    float mx = -3.402823466e38f;
    for (int m = tid; m < M; m += 256) mx = fmaxf(mx, base[(int64_t)m * N]);
    sred[tid] = mx; __syncthreads();
    for (int s = 128; s > 0; s >>= 1) {
        if (tid < s) sred[tid] = fmaxf(sred[tid], sred[tid + s]);
        __syncthreads();
    }
    mx = sred[0]; __syncthreads();
    float sum = 0.0f;
    for (int m = tid; m < M; m += 256) {
        float e = __expf(base[(int64_t)m * N] - mx);
        base[(int64_t)m * N] = e;
        sum += e;
    }
    sred[tid] = sum; __syncthreads();
    for (int s = 128; s > 0; s >>= 1) {
        if (tid < s) sred[tid] += sred[tid + s];
        __syncthreads();
    }
    const float inv = 1.0f / sred[0];
    for (int m = tid; m < M; m += 256) base[(int64_t)m * N] *= inv;
}

// Bilinear resize (reference coordinate convention) with optional fused add.
__global__ void bilinear_kernel(const float* __restrict__ in, float* __restrict__ out,
                                const float* __restrict__ add, int add_bcast,
                                int B, int C, int IH, int IW, int OH, int OW)
{
    const int64_t idx = (int64_t)blockIdx.x * blockDim.x + threadIdx.x;
    const int64_t total = (int64_t)B * C * OH * OW;
    if (idx >= total) return;
    int ow = (int)(idx % OW); int64_t t = idx / OW;
    int oh = (int)(t % OH); t /= OH;
    int c  = (int)(t % C);  int b = (int)(t / C);
    float cy = fmaxf((oh + 0.5f) * ((float)IH / (float)OH) - 0.5f, 0.0f);
    float cx = fmaxf((ow + 0.5f) * ((float)IW / (float)OW) - 0.5f, 0.0f);
    int y0 = (int)floorf(cy); int y1 = y0 + 1 < IH ? y0 + 1 : IH - 1; float wy = cy - (float)y0;
    int x0 = (int)floorf(cx); int x1 = x0 + 1 < IW ? x0 + 1 : IW - 1; float wx = cx - (float)x0;
    const float* p = in + ((int64_t)b * C + c) * IH * IW;
    float v = (p[(int64_t)y0 * IW + x0] * (1.0f - wx) + p[(int64_t)y0 * IW + x1] * wx) * (1.0f - wy)
            + (p[(int64_t)y1 * IW + x0] * (1.0f - wx) + p[(int64_t)y1 * IW + x1] * wx) * wy;
    if (add) {
        const int ab = add_bcast ? 0 : b;
        v += add[(((int64_t)ab * C + c) * OH + oh) * OW + ow];
    }
    out[idx] = v;
}

// mem_out = concat([mem (B,Cm,S), broadcast(qv (1,Cq,S))], channel axis)
__global__ void concat_kernel(const float* __restrict__ mem, const float* __restrict__ qv,
                              float* __restrict__ out, int B, int Cm, int Cq, int S)
{
    const int64_t idx = (int64_t)blockIdx.x * blockDim.x + threadIdx.x;
    const int Ct = Cm + Cq;
    const int64_t total = (int64_t)B * Ct * S;
    if (idx >= total) return;
    int s = (int)(idx % S); int64_t t = idx / S;
    int c = (int)(t % Ct);  int b = (int)(t / Ct);
    out[idx] = (c < Cm) ? mem[((int64_t)b * Cm + c) * S + s]
                        : qv[(int64_t)(c - Cm) * S + s];
}

// diff[b,s] = p2[b,1,s] - p2[b,0,s]  (softmax of 2 channels == sigmoid of diff)
__global__ void chdiff_kernel(const float* __restrict__ p2, float* __restrict__ diff, int B, int S)
{
    const int64_t idx = (int64_t)blockIdx.x * blockDim.x + threadIdx.x;
    if (idx >= (int64_t)B * S) return;
    const int b = (int)(idx / S);
    const int s = (int)(idx % S);
    diff[idx] = p2[((int64_t)b * 2 + 1) * S + s] - p2[(int64_t)b * 2 * S + s];
}

__device__ __forceinline__ float logit_clip(float x)
{
    x = fminf(fmaxf(x, 1e-7f), 1.0f - 1e-7f);
    return __logf(x / (1.0f - x));
}

// For each output pixel: bilinear-upsample diff (x4), sigmoid -> ps[b],
// em0 = prod_b (1-ps[b]); emit logit([em0, ps0..ps3]) into (1,5,OH,OW).
__global__ void final_kernel(const float* __restrict__ diff, float* __restrict__ out,
                             int NB, int IH, int IW, int OH, int OW)
{
    const int64_t p = (int64_t)blockIdx.x * blockDim.x + threadIdx.x;
    const int64_t npix = (int64_t)OH * OW;
    if (p >= npix) return;
    const int ow = (int)(p % OW);
    const int oh = (int)(p / OW);
    float cy = fmaxf((oh + 0.5f) * ((float)IH / (float)OH) - 0.5f, 0.0f);
    float cx = fmaxf((ow + 0.5f) * ((float)IW / (float)OW) - 0.5f, 0.0f);
    int y0 = (int)floorf(cy); int y1 = y0 + 1 < IH ? y0 + 1 : IH - 1; float wy = cy - (float)y0;
    int x0 = (int)floorf(cx); int x1 = x0 + 1 < IW ? x0 + 1 : IW - 1; float wx = cx - (float)x0;
    float em0 = 1.0f;
    float ps[4];
    for (int b = 0; b < 4; ++b) {
        const float* pb = diff + (int64_t)b * IH * IW;
        float v = (pb[(int64_t)y0 * IW + x0] * (1.0f - wx) + pb[(int64_t)y0 * IW + x1] * wx) * (1.0f - wy)
                + (pb[(int64_t)y1 * IW + x0] * (1.0f - wx) + pb[(int64_t)y1 * IW + x1] * wx) * wy;
        float s = 1.0f / (1.0f + __expf(-v));
        ps[b] = s;
        em0 *= (1.0f - s);
    }
    out[p] = logit_clip(em0);
    for (int b = 0; b < 4; ++b) out[(int64_t)(b + 1) * npix + p] = logit_clip(ps[b]);
}

// ---------------------------------------------------------------------------
// Host-side helpers
// ---------------------------------------------------------------------------
static void launch_conv(hipStream_t stream,
                        const float* in, const __bf16* w, const float* b,
                        const float* res, int res_bcast, float* out,
                        int B, int Cin, int IH, int IW, int Cout,
                        int ks, int stride, int relu_in)
{
    const int pad = (ks - 1) / 2;
    const int OH = (IH + 2 * pad - ks) / stride + 1;
    const int OW = (IW + 2 * pad - ks) / stride + 1;
    const int Mtot = B * OH * OW;
    dim3 grid((Mtot + 255) / 256, (Cout + 63) / 64, 1);
    dim3 block(256, 1, 1);
    if (ks == 1)
        conv_wmma_kernel<1, 1><<<grid, block, 0, stream>>>(in, w, b, res, out,
            B, Cin, IH, IW, Cout, OH, OW, stride, pad, relu_in, res_bcast);
    else if (ks == 3)
        conv_wmma_kernel<3, 3><<<grid, block, 0, stream>>>(in, w, b, res, out,
            B, Cin, IH, IW, Cout, OH, OW, stride, pad, relu_in, res_bcast);
    else
        conv_wmma_kernel<7, 7><<<grid, block, 0, stream>>>(in, w, b, res, out,
            B, Cin, IH, IW, Cout, OH, OW, stride, pad, relu_in, res_bcast);
}

static void launch_bilinear(hipStream_t stream, const float* in, float* out,
                            const float* add, int add_bcast,
                            int B, int C, int IH, int IW, int OH, int OW)
{
    const int64_t total = (int64_t)B * C * OH * OW;
    const int grid = (int)((total + 255) / 256);
    bilinear_kernel<<<grid, 256, 0, stream>>>(in, out, add, add_bcast, B, C, IH, IW, OH, OW);
}

// Parameter indices matching _PSPECS order (w at d_in[5+2i], b at d_in[6+2i]).
enum {
    P_ENC_POS = 0, P_ENC_RB1_C1, P_ENC_RB1_C2, P_ENC_RB1_DS,
    P_ENC_RB2_C1, P_ENC_RB2_C2, P_ENC_RB2_DS,
    P_ATN_KEY, P_ATN_VALUE,
    P_QRY_RB1_C1, P_QRY_RB1_C2, P_QRY_RB1_DS,
    P_QRY_RB2_C1, P_QRY_RB2_C2, P_QRY_RB2_DS,
    P_QRY_ATN_KEY, P_QRY_ATN_VALUE,
    P_CONVFM, P_RESMM_C1, P_RESMM_C2,
    P_RF3_CONVFS, P_RF3_RESFS_C1, P_RF3_RESFS_C2, P_RF3_RESMM_C1, P_RF3_RESMM_C2,
    P_RF2_CONVFS, P_RF2_RESFS_C1, P_RF2_RESFS_C2, P_RF2_RESMM_C1, P_RF2_RESMM_C2,
    P_PRED2, P_COUNT
};

// (Cout, Cin, k) per spec, in _PSPECS order.
static const int g_spec_o[P_COUNT] = {256,256,256,256,512,512,512,64,256,256,256,256,
                                      512,512,512,64,256,256,256,256,256,256,256,256,
                                      256,256,256,256,256,256,2};
static const int g_spec_i[P_COUNT] = {1,256,256,256,256,512,256,512,512,256,256,256,
                                      256,512,256,512,512,512,256,256,512,256,256,256,
                                      256,256,256,256,256,256,256};
static const int g_spec_k[P_COUNT] = {7,3,3,3,3,3,3,1,1,3,3,3,3,3,3,1,1,3,3,3,3,3,3,3,
                                      3,3,3,3,3,3,3};

extern "C" void kernel_launch(void* const* d_in, const int* in_sizes, int n_in,
                              void* d_out, int out_size, void* d_ws, size_t ws_size,
                              hipStream_t stream)
{
    (void)in_sizes; (void)n_in; (void)out_size; (void)ws_size;

    const float* ref_x     = (const float*)d_in[0]; // (1,256,96,160)
    const float* x_in      = (const float*)d_in[1]; // (1,256,96,160)
    const float* C2        = (const float*)d_in[2]; // (1,256,96,160)
    const float* C3        = (const float*)d_in[3]; // (1,512,48,80)
    const float* ref_masks = (const float*)d_in[4]; // (4,384,640)
    const float* Wp[P_COUNT];
    const float* Bp[P_COUNT];
    for (int i = 0; i < P_COUNT; ++i) {
        Wp[i] = (const float*)d_in[5 + 2 * i];
        Bp[i] = (const float*)d_in[6 + 2 * i];
    }
    float* ws  = (float*)d_ws;
    float* out = (float*)d_out;

    // Workspace layout (float elements); bf16 weight arena appended at end.
    const size_t OFF_M    = 0;                       //   61,440  (4,1,96,160) mask
    const size_t OFF_BIG0 = OFF_M    + 61440;        // 15,728,640 xenc -> mm2/m2
    const size_t OFF_BIG1 = OFF_BIG0 + 15728640;     // 15,728,640 RF2 temp
    const size_t OFF_MED0 = OFF_BIG1 + 15728640;     //  3,932,160
    const size_t OFF_MED1 = OFF_MED0 + 3932160;      //  3,932,160
    const size_t OFF_MED2 = OFF_MED1 + 3932160;      //  3,932,160
    const size_t OFF_MED3 = OFF_MED2 + 3932160;      //  3,932,160
    const size_t OFF_ATT  = OFF_MED3 + 3932160;      //  3,686,400 (4,960,960)
    const size_t OFF_SMA  = OFF_ATT  + 3686400;      //  1,966,080 enc ds2/h2
    const size_t OFF_SMB  = OFF_SMA  + 1966080;      //  1,966,080 enc t2; qry ds2/h2
    const size_t OFF_KEYS = OFF_SMB  + 1966080;      //    245,760 (4,64,960)
    const size_t OFF_VALS = OFF_KEYS + 245760;       //    983,040 (4,256,960)
    const size_t OFF_QT2  = OFF_VALS + 983040;       //    491,520 (1,512,960)
    const size_t OFF_QK   = OFF_QT2  + 491520;       //     61,440 (1,64,960)
    const size_t OFF_QV   = OFF_QK   + 61440;        //    245,760 (1,256,960)
    const size_t OFF_MEM  = OFF_QV   + 245760;       //    983,040 (4,256,960)
    const size_t OFF_MEMO = OFF_MEM  + 983040;       //  1,966,080 (4,512,960)
    const size_t OFF_FM   = OFF_MEMO + 1966080;      //    983,040 fm/m4
    const size_t OFF_RT   = OFF_FM   + 983040;       //    983,040 ResMM temp
    const size_t OFF_P2   = OFF_RT   + 983040;       //    122,880 (4,2,96,160)
    const size_t OFF_DIFF = OFF_P2   + 122880;       //     61,440 (4,96,160)
    const size_t OFF_WB   = OFF_DIFF + 61440;        // bf16 weight arena (~22M bf16)

    // ---- pre-convert all weights to zero-padded bf16 ----
    __bf16* wb_base = (__bf16*)(ws + OFF_WB);
    const __bf16* Wb[P_COUNT];
    {
        size_t cur = 0;
        for (int i = 0; i < P_COUNT; ++i) {
            const int Ktot = g_spec_i[i] * g_spec_k[i] * g_spec_k[i];
            const int Kpad = (Ktot + 31) & ~31;
            Wb[i] = wb_base + cur;
            const int64_t total = (int64_t)g_spec_o[i] * Kpad;
            wcvt_kernel<<<(int)((total + 255) / 256), 256, 0, stream>>>(
                Wp[i], wb_base + cur, g_spec_o[i], Ktot, Kpad);
            cur += (size_t)g_spec_o[i] * Kpad;
        }
    }

    // ---- encoder path (batch 4) ----
    launch_bilinear(stream, ref_masks, ws + OFF_M, nullptr, 0, 4, 1, 384, 640, 96, 160);
    launch_conv(stream, ws + OFF_M, Wb[P_ENC_POS], Bp[P_ENC_POS], ref_x, 1,
                ws + OFF_BIG0, 4, 1, 96, 160, 256, 7, 1, 0);             // xenc
    launch_conv(stream, ws + OFF_BIG0, Wb[P_ENC_RB1_DS], Bp[P_ENC_RB1_DS], nullptr, 0,
                ws + OFF_MED0, 4, 256, 96, 160, 256, 3, 2, 0);           // ds1
    launch_conv(stream, ws + OFF_BIG0, Wb[P_ENC_RB1_C1], Bp[P_ENC_RB1_C1], nullptr, 0,
                ws + OFF_MED1, 4, 256, 96, 160, 256, 3, 2, 1);           // t1
    launch_conv(stream, ws + OFF_MED1, Wb[P_ENC_RB1_C2], Bp[P_ENC_RB1_C2], ws + OFF_MED0, 0,
                ws + OFF_MED2, 4, 256, 48, 80, 256, 3, 1, 1);            // h1
    launch_conv(stream, ws + OFF_MED2, Wb[P_ENC_RB2_DS], Bp[P_ENC_RB2_DS], nullptr, 0,
                ws + OFF_SMA, 4, 256, 48, 80, 512, 3, 2, 0);             // ds2
    launch_conv(stream, ws + OFF_MED2, Wb[P_ENC_RB2_C1], Bp[P_ENC_RB2_C1], nullptr, 0,
                ws + OFF_SMB, 4, 256, 48, 80, 512, 3, 2, 1);             // t2
    launch_conv(stream, ws + OFF_SMB, Wb[P_ENC_RB2_C2], Bp[P_ENC_RB2_C2], ws + OFF_SMA, 0,
                ws + OFF_SMA, 4, 512, 24, 40, 512, 3, 1, 1);             // h2 (in-place res)
    launch_conv(stream, ws + OFF_SMA, Wb[P_ATN_KEY], Bp[P_ATN_KEY], nullptr, 0,
                ws + OFF_KEYS, 4, 512, 24, 40, 64, 1, 1, 0);
    launch_conv(stream, ws + OFF_SMA, Wb[P_ATN_VALUE], Bp[P_ATN_VALUE], nullptr, 0,
                ws + OFF_VALS, 4, 512, 24, 40, 256, 1, 1, 0);

    // ---- query path (batch 1) ----
    launch_conv(stream, x_in, Wb[P_QRY_RB1_DS], Bp[P_QRY_RB1_DS], nullptr, 0,
                ws + OFF_MED0, 1, 256, 96, 160, 256, 3, 2, 0);
    launch_conv(stream, x_in, Wb[P_QRY_RB1_C1], Bp[P_QRY_RB1_C1], nullptr, 0,
                ws + OFF_MED1, 1, 256, 96, 160, 256, 3, 2, 1);
    launch_conv(stream, ws + OFF_MED1, Wb[P_QRY_RB1_C2], Bp[P_QRY_RB1_C2], ws + OFF_MED0, 0,
                ws + OFF_MED0, 1, 256, 48, 80, 256, 3, 1, 1);            // q_h1 in-place
    launch_conv(stream, ws + OFF_MED0, Wb[P_QRY_RB2_DS], Bp[P_QRY_RB2_DS], nullptr, 0,
                ws + OFF_SMB, 1, 256, 48, 80, 512, 3, 2, 0);             // q_ds2
    launch_conv(stream, ws + OFF_MED0, Wb[P_QRY_RB2_C1], Bp[P_QRY_RB2_C1], nullptr, 0,
                ws + OFF_QT2, 1, 256, 48, 80, 512, 3, 2, 1);             // q_t2
    launch_conv(stream, ws + OFF_QT2, Wb[P_QRY_RB2_C2], Bp[P_QRY_RB2_C2], ws + OFF_SMB, 0,
                ws + OFF_SMB, 1, 512, 24, 40, 512, 3, 1, 1);             // q_h2 in-place
    launch_conv(stream, ws + OFF_SMB, Wb[P_QRY_ATN_KEY], Bp[P_QRY_ATN_KEY], nullptr, 0,
                ws + OFF_QK, 1, 512, 24, 40, 64, 1, 1, 0);
    launch_conv(stream, ws + OFF_SMB, Wb[P_QRY_ATN_VALUE], Bp[P_QRY_ATN_VALUE], nullptr, 0,
                ws + OFF_QV, 1, 512, 24, 40, 256, 1, 1, 0);

    // ---- attention ----
    {
        dim3 grid((960 + 127) / 128, (960 + 63) / 64, 4);
        gemm_wmma_kernel<<<grid, 256, 0, stream>>>(
            ws + OFF_KEYS, ws + OFF_QK, ws + OFF_ATT,
            960, 960, 64,
            /*a_ms=*/1, /*a_ks=*/960, /*b_ks=*/960, /*b_ns=*/1,
            /*a_batch=*/61440, /*b_batch=*/0, /*c_batch=*/921600, 0.125f);
    }
    {
        dim3 grid(960, 4, 1);
        softmax_col_kernel<<<grid, 256, 0, stream>>>(ws + OFF_ATT, 960, 960);
    }
    {
        dim3 grid((256 + 127) / 128, (960 + 63) / 64, 4);
        gemm_wmma_kernel<<<grid, 256, 0, stream>>>(
            ws + OFF_VALS, ws + OFF_ATT, ws + OFF_MEM,
            256, 960, 960,
            /*a_ms=*/960, /*a_ks=*/1, /*b_ks=*/960, /*b_ns=*/1,
            /*a_batch=*/245760, /*b_batch=*/921600, /*c_batch=*/245760, 1.0f);
    }
    {
        const int64_t total = (int64_t)4 * 512 * 960;
        concat_kernel<<<(int)((total + 255) / 256), 256, 0, stream>>>(
            ws + OFF_MEM, ws + OFF_QV, ws + OFF_MEMO, 4, 256, 256, 960);
    }

    // ---- m4 = resblock(convFM(mem_out)) ----
    launch_conv(stream, ws + OFF_MEMO, Wb[P_CONVFM], Bp[P_CONVFM], nullptr, 0,
                ws + OFF_FM, 4, 512, 24, 40, 256, 3, 1, 0);              // fm
    launch_conv(stream, ws + OFF_FM, Wb[P_RESMM_C1], Bp[P_RESMM_C1], nullptr, 0,
                ws + OFF_RT, 4, 256, 24, 40, 256, 3, 1, 1);
    launch_conv(stream, ws + OFF_RT, Wb[P_RESMM_C2], Bp[P_RESMM_C2], ws + OFF_FM, 0,
                ws + OFF_FM, 4, 256, 24, 40, 256, 3, 1, 1);              // m4 in-place

    // ---- RF3 refine (48x80) ----
    launch_conv(stream, C3, Wb[P_RF3_CONVFS], Bp[P_RF3_CONVFS], nullptr, 0,
                ws + OFF_MED1, 1, 512, 48, 80, 256, 3, 1, 0);            // s3a
    launch_conv(stream, ws + OFF_MED1, Wb[P_RF3_RESFS_C1], Bp[P_RF3_RESFS_C1], nullptr, 0,
                ws + OFF_MED2, 1, 256, 48, 80, 256, 3, 1, 1);
    launch_conv(stream, ws + OFF_MED2, Wb[P_RF3_RESFS_C2], Bp[P_RF3_RESFS_C2], ws + OFF_MED1, 0,
                ws + OFF_MED1, 1, 256, 48, 80, 256, 3, 1, 1);            // s3 in-place
    launch_bilinear(stream, ws + OFF_FM, ws + OFF_MED2, ws + OFF_MED1, 1,
                    4, 256, 24, 40, 48, 80);                             // mm3
    launch_conv(stream, ws + OFF_MED2, Wb[P_RF3_RESMM_C1], Bp[P_RF3_RESMM_C1], nullptr, 0,
                ws + OFF_MED3, 4, 256, 48, 80, 256, 3, 1, 1);
    launch_conv(stream, ws + OFF_MED3, Wb[P_RF3_RESMM_C2], Bp[P_RF3_RESMM_C2], ws + OFF_MED2, 0,
                ws + OFF_MED2, 4, 256, 48, 80, 256, 3, 1, 1);            // m3 in-place

    // ---- RF2 refine (96x160) ----
    launch_conv(stream, C2, Wb[P_RF2_CONVFS], Bp[P_RF2_CONVFS], nullptr, 0,
                ws + OFF_MED0, 1, 256, 96, 160, 256, 3, 1, 0);           // s2a
    launch_conv(stream, ws + OFF_MED0, Wb[P_RF2_RESFS_C1], Bp[P_RF2_RESFS_C1], nullptr, 0,
                ws + OFF_MED1, 1, 256, 96, 160, 256, 3, 1, 1);
    launch_conv(stream, ws + OFF_MED1, Wb[P_RF2_RESFS_C2], Bp[P_RF2_RESFS_C2], ws + OFF_MED0, 0,
                ws + OFF_MED0, 1, 256, 96, 160, 256, 3, 1, 1);           // s2 in-place
    launch_bilinear(stream, ws + OFF_MED2, ws + OFF_BIG0, ws + OFF_MED0, 1,
                    4, 256, 48, 80, 96, 160);                            // mm2
    launch_conv(stream, ws + OFF_BIG0, Wb[P_RF2_RESMM_C1], Bp[P_RF2_RESMM_C1], nullptr, 0,
                ws + OFF_BIG1, 4, 256, 96, 160, 256, 3, 1, 1);
    launch_conv(stream, ws + OFF_BIG1, Wb[P_RF2_RESMM_C2], Bp[P_RF2_RESMM_C2], ws + OFF_BIG0, 0,
                ws + OFF_BIG0, 4, 256, 96, 160, 256, 3, 1, 1);           // m2 in-place

    // ---- prediction head ----
    launch_conv(stream, ws + OFF_BIG0, Wb[P_PRED2], Bp[P_PRED2], nullptr, 0,
                ws + OFF_P2, 4, 256, 96, 160, 2, 3, 1, 1);               // p2
    {
        const int64_t total = (int64_t)4 * 96 * 160;
        chdiff_kernel<<<(int)((total + 255) / 256), 256, 0, stream>>>(
            ws + OFF_P2, ws + OFF_DIFF, 4, 96 * 160);
    }
    {
        const int64_t npix = (int64_t)384 * 640;
        final_kernel<<<(int)((npix + 255) / 256), 256, 0, stream>>>(
            ws + OFF_DIFF, out, 4, 96, 160, 384, 640);
    }
}
